// Feedback_Net_23235773071581
// MI455X (gfx1250) — matrix operand order repaired and checked by reading, not compiled, not run on hardware
//
#include <hip/hip_runtime.h>
#include <hip/hip_bf16.h>
#include <stdint.h>

typedef __attribute__((ext_vector_type(16))) _Float16 v16h;
typedef __attribute__((ext_vector_type(8)))  float    v8f;
typedef __attribute__((ext_vector_type(8)))  uint32_t v8u;
typedef __attribute__((ext_vector_type(4)))  unsigned int u32x4;
typedef __attribute__((ext_vector_type(8)))  int      i32x8;
typedef __attribute__((ext_vector_type(4)))  int      i32x4;

#define T_SEQ   128
#define EMB_D   128
#define HID     64
#define BATCH_N 512
#define NROW    (BATCH_N * T_SEQ)   // 65536 (b*T + t) rows
#define NTAGS   9

// ---------------------------------------------------------------------------
// WMMA helpers (CDNA5 wave32, 16x16x32 f16 -> f32)
// ---------------------------------------------------------------------------
__device__ __forceinline__ v8f wmma16(v16h a, v16h b, v8f c) {
  return __builtin_amdgcn_wmma_f32_16x16x32_f16(false, a, false, b,
                                                (short)0, c, false, false);
}

// A fragment (16x32, MxK f16). rowptr = &A[m][0] for this lane's row m (=lane&15).
__device__ __forceinline__ v16h load_a_frag(const _Float16* __restrict__ rowptr,
                                            int kh, int kbase) {
  v8u r;
#pragma unroll
  for (int v = 0; v < 8; ++v) {
    int k0 = kbase + ((v < 4) ? (kh * 8 + 2 * v) : (16 + kh * 8 + 2 * (v - 4)));
    r[v] = *(const uint32_t*)(rowptr + k0);
  }
  return __builtin_bit_cast(v16h, r);
}

// B fragment (32x16, KxN f16) from n-major staged weights wT[N][ldk].
__device__ __forceinline__ v16h load_b_frag(const _Float16* __restrict__ wT,
                                            int n, int kh, int kbase, int ldk) {
  const _Float16* p = wT + (size_t)n * ldk + kbase + kh * 8;
  v8u r;
#pragma unroll
  for (int j = 0; j < 8; ++j) r[j] = *(const uint32_t*)(p + ((j < 4) ? 2 * j : 16 + 2 * (j - 4)));
  return __builtin_bit_cast(v16h, r);
}

__device__ __forceinline__ float fast_sigmoid(float z) {
  return __builtin_amdgcn_rcpf(1.0f + __expf(-z));
}
__device__ __forceinline__ float fast_tanh(float z) {
  return 2.0f * __builtin_amdgcn_rcpf(1.0f + __expf(-2.0f * z)) - 1.0f;
}

// ---------------------------------------------------------------------------
// Kernel 1: embedding lookup + LayerNorm -> f16 activations [NROW][128]
// ---------------------------------------------------------------------------
__global__ void k_embed_ln(const int* __restrict__ X, const float* __restrict__ emb,
                           const float* __restrict__ gamma, const float* __restrict__ beta,
                           _Float16* __restrict__ x0) {
  const int row  = blockIdx.x * (blockDim.x >> 5) + (threadIdx.x >> 5);
  const int lane = threadIdx.x & 31;
  if (row >= NROW) return;
  const int tok = X[row];
  const float4 v = *(const float4*)(emb + (size_t)tok * EMB_D + lane * 4);
  float s  = v.x + v.y + v.z + v.w;
  float s2 = v.x * v.x + v.y * v.y + v.z * v.z + v.w * v.w;
#pragma unroll
  for (int m = 16; m >= 1; m >>= 1) {
    s  += __shfl_xor(s,  m, 32);
    s2 += __shfl_xor(s2, m, 32);
  }
  const float mu  = s * (1.0f / 128.0f);
  const float var = s2 * (1.0f / 128.0f) - mu * mu;
  const float rs  = rsqrtf(var + 1e-3f);
  const float4 g  = *(const float4*)(gamma + lane * 4);
  const float4 be = *(const float4*)(beta + lane * 4);
  union { _Float16 h[4]; uint2 u; } pk;
  pk.h[0] = (_Float16)((v.x - mu) * rs * g.x + be.x);
  pk.h[1] = (_Float16)((v.y - mu) * rs * g.y + be.y);
  pk.h[2] = (_Float16)((v.z - mu) * rs * g.z + be.z);
  pk.h[3] = (_Float16)((v.w - mu) * rs * g.w + be.w);
  *(uint2*)(x0 + (size_t)row * EMB_D + lane * 4) = pk.u;
}

// ---------------------------------------------------------------------------
// Kernel 2: stage weights f32 -> f16, n-major for B fragments.
// ---------------------------------------------------------------------------
__global__ void k_stage_w(const float* __restrict__ Wx, const float* __restrict__ Wh,
                          _Float16* __restrict__ WxT, _Float16* __restrict__ WhT) {
  const int idx = blockIdx.x * blockDim.x + threadIdx.x;
  const int NX = 3 * 2 * 256 * 128;
  const int NH = 3 * 2 * 256 * 64;
  if (idx < NX) {
    const int k = idx & 127, n = (idx >> 7) & 255, ld = idx >> 15;
    WxT[idx] = (_Float16)Wx[((size_t)ld * 128 + k) * 256 + n];
  } else if (idx < NX + NH) {
    const int j = idx - NX;
    const int k = j & 63, n = (j >> 6) & 255, ld = j >> 14;
    WhT[j] = (_Float16)Wh[((size_t)ld * 64 + k) * 256 + n];
  }
}

// ---------------------------------------------------------------------------
// Kernel 3: input projection, both dirs. M=32 per wave (2 M-tiles) so every
// B fragment feeds 2 WMMAs.  Output gx is TIME-MAJOR: gx[t][b][dir*256+n] f16.
// 32 static v_wmma per wave.
// ---------------------------------------------------------------------------
__global__ void k_xgemm(const _Float16* __restrict__ xin, const _Float16* __restrict__ WxTl,
                        const float* __restrict__ biasl, _Float16* __restrict__ gx) {
  const int dir  = blockIdx.y;
  const int mt   = blockIdx.x;              // 32 rows of (b*T+t)
  const int wave = threadIdx.x >> 5;
  const int lane = threadIdx.x & 31;
  const int m = lane & 15, kh = lane >> 4;

  const _Float16* arow0 = xin + (size_t)(mt * 32 + m) * EMB_D;
  const _Float16* arow1 = arow0 + (size_t)16 * EMB_D;
  const _Float16* wT    = WxTl + (size_t)dir * 256 * EMB_D;

  v16h a0[4], a1[4];
#pragma unroll
  for (int s = 0; s < 4; ++s) {
    a0[s] = load_a_frag(arow0, kh, s * 32);
    a1[s] = load_a_frag(arow1, kh, s * 32);
  }

#pragma unroll
  for (int q = 0; q < 4; ++q) {
    const int n0 = wave * 64 + q * 16;
    v8f acc0, acc1;
#pragma unroll
    for (int j = 0; j < 8; ++j) { acc0[j] = 0.0f; acc1[j] = 0.0f; }
#pragma unroll
    for (int s = 0; s < 4; ++s) {
      const v16h bf = load_b_frag(wT, n0 + m, kh, s * 32, EMB_D);
      acc0 = wmma16(a0[s], bf, acc0);
      acc1 = wmma16(a1[s], bf, acc1);
    }
    const float bv = biasl[dir * 256 + n0 + m];
#pragma unroll
    for (int tile = 0; tile < 2; ++tile) {
      const v8f acc = tile ? acc1 : acc0;
#pragma unroll
      for (int j = 0; j < 8; ++j) {
        const int r  = mt * 32 + tile * 16 + kh * 8 + j;  // (b*T + t)
        const int bb = r >> 7, tt = r & 127;
        gx[(size_t)tt * (BATCH_N * 512) + (size_t)bb * 512 + dir * 256 + n0 + m] =
            (_Float16)(acc[j] + bv);
      }
    }
  }
}

// ---------------------------------------------------------------------------
// Kernel 4: sequential LSTM, 16-row batch tile, one direction, grid (32,2),
// block 128 (wave w = gate w).  Wh fragments stay in VGPRs for all 128 steps.
// Per-step gate tile (16x256 f16, row stride 512 elems) is double-buffered
// into LDS by the Tensor Data Mover, issued 2 steps ahead, synced with
// s_wait_tensorcnt + workgroup barriers.
// ---------------------------------------------------------------------------
__global__ void k_lstm(const _Float16* __restrict__ gx, const _Float16* __restrict__ WhTl,
                       _Float16* __restrict__ xout) {
  __shared__ _Float16 hbuf[16][HID + 8];
  __shared__ float    gates[16][256 + 4];
  __shared__ float    cbuf[16][HID];
  __shared__ _Float16 gstage[2][16][256];   // TDM landing buffers (2 x 8KB)

  const int dir  = blockIdx.y;
  const int m0   = blockIdx.x * 16;
  const int tid  = threadIdx.x;
  const int wave = tid >> 5;
  const int lane = tid & 31;
  const int m = lane & 15, kh = lane >> 4;

  // ---- TDM descriptor issue: 2D tile 256 x 16 of f16, row stride 512 elems.
  auto issue_tdm = [&](int tlog, int bufidx) {
    const int ts = dir ? (T_SEQ - 1 - tlog) : tlog;
    const uint64_t ga = (uint64_t)(uintptr_t)gx +
        2ull * ((size_t)ts * (BATCH_N * 512) + (size_t)m0 * 512 + (size_t)dir * 256);
    const uint32_t lds = (uint32_t)(uintptr_t)&gstage[bufidx][0][0];
    u32x4 g0;
    g0.x = 1u;                                  // count=1 (valid), user mode
    g0.y = lds;                                 // lds_addr
    g0.z = (uint32_t)ga;                        // global_addr[31:0]
    g0.w = (uint32_t)(ga >> 32) | (2u << 30);   // global_addr[56:32] | type=2
    i32x8 g1;
    g1[0] = 0x10000;                            // data_size = 2 bytes
    g1[1] = (int)(256u << 16);                  // tensor_dim0 = 256
    g1[2] = (int)(16u  << 16);                  // tensor_dim1 = 16
    g1[3] = (int)(256u << 16);                  // tile_dim0 = 256
    g1[4] = 16;                                 // tile_dim1 = 16
    g1[5] = 512;                                // tensor_dim0_stride = 512
    g1[6] = 0;
    g1[7] = 0;                                  // dim1 stride unused
    i32x4 z4;
    z4[0] = 0; z4[1] = 0; z4[2] = 0; z4[3] = 0;
    i32x8 z8;
#pragma unroll
    for (int i = 0; i < 8; ++i) z8[i] = 0;
    // amdgpu-toolchain (clang-23) 6-arg form; extra groups unused for 2D tiles
    __builtin_amdgcn_tensor_load_to_lds(g0, g1, z4, z4, z8, 0);
  };

  // Hoist Wh B fragments (this wave's gate, K=64) into VGPRs.
  const _Float16* wT = WhTl + (size_t)dir * 256 * HID;
  v16h bfr[4][2];
#pragma unroll
  for (int q = 0; q < 4; ++q)
#pragma unroll
    for (int s = 0; s < 2; ++s)
      bfr[q][s] = load_b_frag(wT, wave * 64 + q * 16 + m, kh, s * 32, HID);

  if (wave == 0) {                 // prime the pipeline: tiles for t=0 and t=1
    issue_tdm(0, 0);
    issue_tdm(1, 1);
    __builtin_amdgcn_s_wait_tensorcnt(1);   // tile 0 landed
  }

  // h = 0, c = 0
#pragma unroll
  for (int i = 0; i < 8; ++i) {
    const int e = tid * 8 + i, mm = e >> 6, nn = e & 63;
    hbuf[mm][nn] = (_Float16)0.0f;
    cbuf[mm][nn] = 0.0f;
  }
  __syncthreads();

  const int gate = wave;
  for (int t = 0; t < T_SEQ; ++t) {
    const int tt  = dir ? (T_SEQ - 1 - t) : t;
    const int cur = t & 1;

    // ---- phase 1: C init from staged tile, A from h, WMMA, activations
    v8f acc[4];
#pragma unroll
    for (int q = 0; q < 4; ++q)
#pragma unroll
      for (int j = 0; j < 8; ++j)
        acc[q][j] = (float)gstage[cur][kh * 8 + j][wave * 64 + q * 16 + m];

    v16h a[2];
#pragma unroll
    for (int s = 0; s < 2; ++s) {
      v8u r;
#pragma unroll
      for (int v = 0; v < 8; ++v) {
        const int k0 = s * 32 + ((v < 4) ? (kh * 8 + 2 * v) : (16 + kh * 8 + 2 * (v - 4)));
        r[v] = *(const uint32_t*)&hbuf[m][k0];
      }
      a[s] = __builtin_bit_cast(v16h, r);
    }

#pragma unroll
    for (int q = 0; q < 4; ++q) {
      acc[q] = wmma16(a[0], bfr[q][0], acc[q]);
      acc[q] = wmma16(a[1], bfr[q][1], acc[q]);
    }

#pragma unroll
    for (int q = 0; q < 4; ++q)
#pragma unroll
      for (int j = 0; j < 8; ++j) {
        const float z  = acc[q][j];
        const float av = (gate == 2) ? fast_tanh(z) : fast_sigmoid(z);
        gates[kh * 8 + j][gate * 64 + q * 16 + m] = av;
      }
    __syncthreads();   // barrier 1: all gstage[cur]/hbuf reads + gate writes done

    // ---- TDM: refill the buffer just consumed with the t+2 tile; guarantee
    //      t+1's tile has landed before barrier 2 releases the next step.
    if (wave == 0) {
      if (t + 2 < T_SEQ) {
        issue_tdm(t + 2, cur);
        __builtin_amdgcn_s_wait_tensorcnt(1);
      } else {
        __builtin_amdgcn_s_wait_tensorcnt(0);
      }
    }

    // ---- phase 2: c/h elementwise update
    {
      const int base = tid * 8, mm = base >> 6, nn = base & 63;
      float hv[8];
#pragma unroll
      for (int i = 0; i < 8; ++i) {
        const float iv = gates[mm][nn + i];
        const float fv = gates[mm][64 + nn + i];
        const float gv = gates[mm][128 + nn + i];
        const float ov = gates[mm][192 + nn + i];
        const float c  = fv * cbuf[mm][nn + i] + iv * gv;
        cbuf[mm][nn + i] = c;
        hv[i] = ov * fast_tanh(c);
      }
      union { _Float16 h[8]; uint4 u; } pk;
#pragma unroll
      for (int i = 0; i < 8; ++i) pk.h[i] = (_Float16)hv[i];
      *(uint4*)&hbuf[mm][nn] = pk.u;
      *(uint4*)(xout + ((size_t)(m0 + mm) * T_SEQ + tt) * EMB_D + dir * HID + nn) = pk.u;
    }
    __syncthreads();   // barrier 2: h/c committed, staged tile t+1 ready
  }
}

// ---------------------------------------------------------------------------
// Kernel 5: TimeDistributed Dense(9) + softmax
// ---------------------------------------------------------------------------
__global__ void k_dense(const _Float16* __restrict__ x, const float* __restrict__ Wd,
                        const float* __restrict__ bd, float* __restrict__ out) {
  const int row = blockIdx.x * blockDim.x + threadIdx.x;
  if (row >= NROW) return;
  const _Float16* xr = x + (size_t)row * EMB_D;
  float acc[NTAGS];
#pragma unroll
  for (int j = 0; j < NTAGS; ++j) acc[j] = bd[j];
  for (int k = 0; k < EMB_D; ++k) {
    const float xv = (float)xr[k];
#pragma unroll
    for (int j = 0; j < NTAGS; ++j) acc[j] += xv * Wd[k * NTAGS + j];
  }
  float mx = acc[0];
#pragma unroll
  for (int j = 1; j < NTAGS; ++j) mx = fmaxf(mx, acc[j]);
  float s = 0.0f;
#pragma unroll
  for (int j = 0; j < NTAGS; ++j) { acc[j] = __expf(acc[j] - mx); s += acc[j]; }
  const float inv = 1.0f / s;
#pragma unroll
  for (int j = 0; j < NTAGS; ++j) out[(size_t)row * NTAGS + j] = acc[j] * inv;
}

// ---------------------------------------------------------------------------
extern "C" void kernel_launch(void* const* d_in, const int* in_sizes, int n_in,
                              void* d_out, int out_size, void* d_ws, size_t ws_size,
                              hipStream_t stream) {
  const int*   X     = (const int*)d_in[0];
  const float* emb   = (const float*)d_in[1];
  const float* gamma = (const float*)d_in[2];
  const float* beta  = (const float*)d_in[3];
  const float* Wx    = (const float*)d_in[4];
  const float* Wh    = (const float*)d_in[5];
  const float* b     = (const float*)d_in[6];
  const float* Wd    = (const float*)d_in[7];
  const float* bd    = (const float*)d_in[8];
  float* out = (float*)d_out;

  char* ws = (char*)d_ws;
  size_t off = 0;
  auto carve = [&](size_t bytes) -> void* {
    void* p = ws + off;
    off += (bytes + 255) & ~(size_t)255;
    return p;
  };
  _Float16* xA  = (_Float16*)carve((size_t)NROW * EMB_D * 2);   // 16.8 MB
  _Float16* xB  = (_Float16*)carve((size_t)NROW * EMB_D * 2);   // 16.8 MB
  _Float16* gx  = (_Float16*)carve((size_t)NROW * 512 * 2);     // 67.1 MB, [t][b][512]
  _Float16* WxT = (_Float16*)carve((size_t)3 * 2 * 256 * 128 * 2);
  _Float16* WhT = (_Float16*)carve((size_t)3 * 2 * 256 * 64 * 2);

  k_embed_ln<<<NROW / 8, 256, 0, stream>>>(X, emb, gamma, beta, xA);
  k_stage_w<<<(3 * 2 * 256 * (128 + 64) + 255) / 256, 256, 0, stream>>>(Wx, Wh, WxT, WhT);

  _Float16* xin  = xA;
  _Float16* xout = xB;
  for (int l = 0; l < 3; ++l) {
    k_xgemm<<<dim3(NROW / 32, 2), 128, 0, stream>>>(
        xin, WxT + (size_t)l * 2 * 256 * 128, b + (size_t)l * 2 * 256, gx);
    k_lstm<<<dim3(BATCH_N / 16, 2), 128, 0, stream>>>(
        gx, WhT + (size_t)l * 2 * 256 * 64, xout);
    _Float16* tmp = xin; xin = xout; xout = tmp;
  }
  k_dense<<<(NROW + 255) / 256, 256, 0, stream>>>(xin, Wd, bd, out);
}
